// SharedRoutedMoE_52441550684579
// MI455X (gfx1250) — compile-verified, matrix operand order checked by reading, not run on hardware
//
#include <hip/hip_runtime.h>
#include <math.h>

// ---------------- problem constants (match reference) ----------------
#define D_MODEL 1024
#define NEXP    12
#define HS_DIM  2048
#define HR_DIM  3072
#define TAU_F   1.5f

// ---------------- tiling ----------------
#define MT  64     // token rows per block tile (4 WMMA m-tiles)
#define NT  128    // output cols per block tile
#define KT  64     // K chunk = 2 WMMA k-steps
#define LDK 72     // LDS K stride in bf16 halves (144 B, 16B-aligned, skewed)

typedef __attribute__((ext_vector_type(16))) __bf16 v16bf;
typedef __attribute__((ext_vector_type(8)))  __bf16 v8bf;
typedef __attribute__((ext_vector_type(4)))  __bf16 v4bf;
typedef __attribute__((ext_vector_type(8)))  float  v8f;

__device__ __forceinline__ __bf16 f2bf(float f) { return (__bf16)f; }

__device__ __forceinline__ v8f zero8() { return (v8f)(0.0f); }

__device__ __forceinline__ v8f wmma_bf16(v16bf a, v16bf b, v8f c) {
  return __builtin_amdgcn_wmma_f32_16x16x32_bf16(false, a, false, b, (short)0, c,
                                                 false, false);
}

// Fragment load from LDS bf16 tile laid out [row][K] with stride ld halves.
// ISA 7.12.2 16-bit operand pattern: lane half h needs K = h*8+i (i<8) and
// K = h*8+16+(i-8) (i>=8)  ->  two contiguous 16-byte runs -> 2x ds_load_b128.
__device__ __forceinline__ v16bf frag_ld(const __bf16* base, int ld, int row0,
                                         int koff, int lane) {
  const __bf16* p = base + (size_t)(row0 + (lane & 15)) * ld + koff + (lane >> 4) * 8;
  v8bf lo = *(const v8bf*)(p);
  v8bf hi = *(const v8bf*)(p + 16);
  return __builtin_shufflevector(lo, hi, 0, 1, 2, 3, 4, 5, 6, 7,
                                 8, 9, 10, 11, 12, 13, 14, 15);
}

__device__ __forceinline__ float gelu_exact(float g) {
  return 0.5f * g * (1.0f + erff(g * 0.70710678118654752440f));
}

// ---------------- preprocessing: convert / transpose to bf16 ----------------
__global__ __launch_bounds__(256) void cvt_bf16(const float* __restrict__ in,
                                                __bf16* __restrict__ out, int n) {
  int i = (blockIdx.x * 256 + threadIdx.x) * 4;
  if (i >= n) return;
  float4 v = *(const float4*)(in + i);
  v4bf p;
  p[0] = f2bf(v.x); p[1] = f2bf(v.y); p[2] = f2bf(v.z); p[3] = f2bf(v.w);
  *(v4bf*)(out + i) = p;
}

// batched transpose+convert: in f32 [R][C] -> out bf16 [C][R]; R,C % 32 == 0
__global__ __launch_bounds__(256) void transpose_cvt(const float* __restrict__ in,
                                                     __bf16* __restrict__ out,
                                                     int R, int C) {
  __shared__ float t[32][33];
  size_t base = (size_t)blockIdx.z * R * C;
  int r0 = blockIdx.y * 32, c0 = blockIdx.x * 32;
  int lx = threadIdx.x & 31, ly = threadIdx.x >> 5;  // 32 x 8
#pragma unroll
  for (int i = 0; i < 32; i += 8)
    t[ly + i][lx] = in[base + (size_t)(r0 + ly + i) * C + c0 + lx];
  __syncthreads();
#pragma unroll
  for (int i = 0; i < 32; i += 8)
    out[base + (size_t)(c0 + ly + i) * R + r0 + lx] = f2bf(t[lx][ly + i]);
}

// ---------------- router: exact fp32 logits, top-2 softmax, expert lists ---
__global__ __launch_bounds__(256) void moe_router(
    const float* __restrict__ x, const float* __restrict__ Wr,
    const float* __restrict__ br, int* __restrict__ counts,
    int* __restrict__ rowlist, float* __restrict__ wlist, int Ntok) {
  int wv = threadIdx.x >> 5;
  int lane = threadIdx.x & 31;
  int t = blockIdx.x * 8 + wv;
  if (t >= Ntok) return;
  const float* xt = x + (size_t)t * D_MODEL;
  float acc[NEXP];
#pragma unroll
  for (int e = 0; e < NEXP; ++e) acc[e] = 0.0f;
  for (int d = lane; d < D_MODEL; d += 32) {
    float xv = xt[d];
    const float* wr = Wr + (size_t)d * NEXP;
#pragma unroll
    for (int e = 0; e < NEXP; ++e) acc[e] += xv * wr[e];
  }
#pragma unroll
  for (int e = 0; e < NEXP; ++e) {
    float v = acc[e];
#pragma unroll
    for (int off = 16; off > 0; off >>= 1) v += __shfl_xor(v, off, 32);
    acc[e] = v;
  }
  if (lane == 0) {
    float b0 = -3.0e38f, b1 = -3.0e38f;
    int i0 = 0, i1 = 1;
#pragma unroll
    for (int e = 0; e < NEXP; ++e) {
      float l = (acc[e] + br[e]) * (1.0f / TAU_F);
      if (l > b0) { b1 = b0; i1 = i0; b0 = l; i0 = e; }
      else if (l > b1) { b1 = l; i1 = e; }
    }
    float w0 = 1.0f / (1.0f + expf(b1 - b0));
    float w1 = 1.0f - w0;
    int p0 = atomicAdd(&counts[i0], 1);
    rowlist[i0 * Ntok + p0] = 2 * t;
    wlist[i0 * Ntok + p0] = w0;
    int p1 = atomicAdd(&counts[i1], 1);
    rowlist[i1 * Ntok + p1] = 2 * t + 1;
    wlist[i1 * Ntok + p1] = w1;
  }
}

// ---------------- layer 1: ACT = GEGLU(X @ W1 + b1), bf16 WMMA --------------
// Double-buffered LDS: global loads for chunk k+1 overlap WMMAs of chunk k.
__global__ __launch_bounds__(256) void moe_l1(
    const __bf16* __restrict__ Xb, const __bf16* __restrict__ W1t,
    const float* __restrict__ b1, __bf16* __restrict__ ACT,
    const int* __restrict__ rowlist, const int* __restrict__ counts,
    int Ntok, int Kdim, int H) {
  const int e = blockIdx.z;
  const bool routed = (rowlist != nullptr);
  const int count = routed ? counts[e] : Ntok;
  const int m0 = blockIdx.y * MT;
  if (m0 >= count) return;  // block-uniform exit
  const int c0 = blockIdx.x * NT;
  const __bf16* Wt = W1t + (routed ? (size_t)e * 2 * H * Kdim : 0);
  const float* bb = b1 + (routed ? (size_t)e * 2 * H : 0);

  __shared__ __align__(16) __bf16 As[2][MT][LDK];
  __shared__ __align__(16) __bf16 Bs[2][2][NT][LDK];
  __shared__ int rows_s[MT];

  const int tid = threadIdx.x;
  if (tid < MT) {
    int m = m0 + tid;
    rows_s[tid] = routed ? ((m < count) ? rowlist[e * Ntok + m] : -1) : m;
  }
  __syncthreads();

  // ---- hoisted staging pointers (K-invariant) ----
  const int ar = tid >> 3;            // A rows: ar and ar+32
  const int aseg = (tid & 7) * 8;
  int rid0 = rows_s[ar], rid1 = rows_s[ar + 32];
  int tok0 = routed ? ((rid0 >= 0) ? (rid0 >> 1) : 0) : rid0;
  int tok1 = routed ? ((rid1 >= 0) ? (rid1 >> 1) : 0) : rid1;
  const __bf16* ag0 = Xb + (size_t)tok0 * Kdim + aseg;
  const __bf16* ag1 = Xb + (size_t)tok1 * Kdim + aseg;
  __bf16* al0 = &As[0][ar][aseg];
  __bf16* al1 = &As[0][ar + 32][aseg];
  const int ABUF = MT * LDK;          // halves per A buffer
  const int BBUF = 2 * NT * LDK;      // halves per B buffer

  const __bf16* bgp[8];
  __bf16* blp[8];
#pragma unroll
  for (int h = 0; h < 2; ++h)
#pragma unroll
    for (int it = 0; it < 4; ++it) {
      int idx = it * 256 + tid;         // 0..1023 16B segments
      int col = idx >> 3, seg = (idx & 7) * 8;
      bgp[h * 4 + it] = Wt + (size_t)(h * H + c0 + col) * Kdim + seg;
      blp[h * 4 + it] = &Bs[0][h][col][seg];
    }

  const int lane = tid & 31, wv = tid >> 5;
  const int mi = wv >> 1;  // 0..3 -> m sub-tile
  const int ni = wv & 1;   // 0..1 -> cols [ni*64, ni*64+64)
  v8f accA[4] = {zero8(), zero8(), zero8(), zero8()};
  v8f accG[4] = {zero8(), zero8(), zero8(), zero8()};

  // prologue: stage chunk 0 into buffer 0
  v8bf ta0 = *(const v8bf*)(ag0);
  v8bf ta1 = *(const v8bf*)(ag1);
  v8bf tb[8];
#pragma unroll
  for (int q = 0; q < 8; ++q) tb[q] = *(const v8bf*)(bgp[q]);
  *(v8bf*)al0 = ta0;
  *(v8bf*)al1 = ta1;
#pragma unroll
  for (int q = 0; q < 8; ++q) *(v8bf*)blp[q] = tb[q];
  __syncthreads();

  const int nk = Kdim / KT;
  for (int k = 0; k < nk; ++k) {
    const int kn = (k + 1) * KT;
    const bool more = kn < Kdim;
    if (more) {  // issue next chunk's global loads (overlap with WMMAs below)
      ta0 = *(const v8bf*)(ag0 + kn);
      ta1 = *(const v8bf*)(ag1 + kn);
#pragma unroll
      for (int q = 0; q < 8; ++q) tb[q] = *(const v8bf*)(bgp[q] + kn);
    }
    const int b = k & 1;
    const __bf16* Ab = &As[b][0][0];
    const __bf16* B0 = &Bs[b][0][0][0];
    const __bf16* B1 = &Bs[b][1][0][0];
#pragma unroll
    for (int s = 0; s < KT; s += 32) {
      v16bf af = frag_ld(Ab, LDK, mi * 16, s, lane);
#pragma unroll
      for (int j = 0; j < 4; ++j) {
        v16bf ba = frag_ld(B0, LDK, ni * 64 + j * 16, s, lane);
        v16bf bg = frag_ld(B1, LDK, ni * 64 + j * 16, s, lane);
        accA[j] = wmma_bf16(af, ba, accA[j]);
        accG[j] = wmma_bf16(af, bg, accG[j]);
      }
    }
    if (more) {  // store staged chunk into the other buffer
      const int o = (b ^ 1) ? 1 : 0;
      *(v8bf*)(al0 + o * ABUF) = ta0;
      *(v8bf*)(al1 + o * ABUF) = ta1;
#pragma unroll
      for (int q = 0; q < 8; ++q) *(v8bf*)(blp[q] + o * BBUF) = tb[q];
    }
    __syncthreads();
  }

  // epilogue: bias + GEGLU, store bf16, scatter rows
  const int half = lane >> 4, ncol = lane & 15;
#pragma unroll
  for (int j = 0; j < 4; ++j) {
    int c = c0 + ni * 64 + j * 16 + ncol;
    float biasA = bb[c], biasG = bb[H + c];
#pragma unroll
    for (int r = 0; r < 8; ++r) {
      int mloc = mi * 16 + half * 8 + r;
      if (m0 + mloc < count) {
        int rid = rows_s[mloc];
        float a = accA[j][r] + biasA;
        float g = accG[j][r] + biasG;
        ACT[(size_t)rid * H + c] = f2bf(a * gelu_exact(g));
      }
    }
  }
}

// ---------------- layer 2: Y = (ACT @ W2 + b2) * w, bf16 WMMA ----------------
__global__ __launch_bounds__(256) void moe_l2(
    const __bf16* __restrict__ ACT, const __bf16* __restrict__ W2t,
    const float* __restrict__ b2, float* __restrict__ Y,
    const int* __restrict__ rowlist, const int* __restrict__ counts,
    const float* __restrict__ wlist, int Ntok, int Kdim, int Dout) {
  const int e = blockIdx.z;
  const bool routed = (rowlist != nullptr);
  const int count = routed ? counts[e] : Ntok;
  const int m0 = blockIdx.y * MT;
  if (m0 >= count) return;
  const int c0 = blockIdx.x * NT;
  const __bf16* Wt = W2t + (routed ? (size_t)e * Dout * Kdim : 0);
  const float* bb = b2 + (routed ? (size_t)e * Dout : 0);

  __shared__ __align__(16) __bf16 As[2][MT][LDK];
  __shared__ __align__(16) __bf16 Bs[2][NT][LDK];
  __shared__ int rows_s[MT];
  __shared__ float ws_s[MT];

  const int tid = threadIdx.x;
  if (tid < MT) {
    int m = m0 + tid;
    if (routed) {
      rows_s[tid] = (m < count) ? rowlist[e * Ntok + m] : -1;
      ws_s[tid]   = (m < count) ? wlist[e * Ntok + m] : 0.0f;
    } else {
      rows_s[tid] = m;
      ws_s[tid]   = 1.0f;
    }
  }
  __syncthreads();

  // ---- hoisted staging pointers ----
  const int ar = tid >> 3;
  const int aseg = (tid & 7) * 8;
  int rid0 = rows_s[ar], rid1 = rows_s[ar + 32];
  int tr0 = (rid0 >= 0) ? rid0 : 0;
  int tr1 = (rid1 >= 0) ? rid1 : 0;
  const __bf16* ag0 = ACT + (size_t)tr0 * Kdim + aseg;
  const __bf16* ag1 = ACT + (size_t)tr1 * Kdim + aseg;
  __bf16* al0 = &As[0][ar][aseg];
  __bf16* al1 = &As[0][ar + 32][aseg];
  const int ABUF = MT * LDK;
  const int BBUF = NT * LDK;

  const __bf16* bgp[4];
  __bf16* blp[4];
#pragma unroll
  for (int it = 0; it < 4; ++it) {
    int idx = it * 256 + tid;
    int col = idx >> 3, seg = (idx & 7) * 8;
    bgp[it] = Wt + (size_t)(c0 + col) * Kdim + seg;
    blp[it] = &Bs[0][col][seg];
  }

  const int lane = tid & 31, wv = tid >> 5;
  const int mi = wv >> 1, ni = wv & 1;
  v8f acc[4] = {zero8(), zero8(), zero8(), zero8()};

  // prologue: stage chunk 0 into buffer 0
  v8bf ta0 = *(const v8bf*)(ag0);
  v8bf ta1 = *(const v8bf*)(ag1);
  v8bf tb[4];
#pragma unroll
  for (int q = 0; q < 4; ++q) tb[q] = *(const v8bf*)(bgp[q]);
  *(v8bf*)al0 = ta0;
  *(v8bf*)al1 = ta1;
#pragma unroll
  for (int q = 0; q < 4; ++q) *(v8bf*)blp[q] = tb[q];
  __syncthreads();

  const int nk = Kdim / KT;
  for (int k = 0; k < nk; ++k) {
    const int kn = (k + 1) * KT;
    const bool more = kn < Kdim;
    if (more) {
      ta0 = *(const v8bf*)(ag0 + kn);
      ta1 = *(const v8bf*)(ag1 + kn);
#pragma unroll
      for (int q = 0; q < 4; ++q) tb[q] = *(const v8bf*)(bgp[q] + kn);
    }
    const int b = k & 1;
    const __bf16* Ab = &As[b][0][0];
    const __bf16* Bb = &Bs[b][0][0];
#pragma unroll
    for (int s = 0; s < KT; s += 32) {
      v16bf af = frag_ld(Ab, LDK, mi * 16, s, lane);
#pragma unroll
      for (int j = 0; j < 4; ++j) {
        v16bf bf = frag_ld(Bb, LDK, ni * 64 + j * 16, s, lane);
        acc[j] = wmma_bf16(af, bf, acc[j]);
      }
    }
    if (more) {
      const int o = (b ^ 1) ? 1 : 0;
      *(v8bf*)(al0 + o * ABUF) = ta0;
      *(v8bf*)(al1 + o * ABUF) = ta1;
#pragma unroll
      for (int q = 0; q < 4; ++q) *(v8bf*)(blp[q] + o * BBUF) = tb[q];
    }
    __syncthreads();
  }

  const int half = lane >> 4, ncol = lane & 15;
#pragma unroll
  for (int j = 0; j < 4; ++j) {
    int c = c0 + ni * 64 + j * 16 + ncol;
    float bias = bb[c];
#pragma unroll
    for (int r = 0; r < 8; ++r) {
      int mloc = mi * 16 + half * 8 + r;
      if (m0 + mloc < count) {
        int orow = rows_s[mloc];
        Y[(size_t)orow * Dout + c] = (acc[j][r] + bias) * ws_s[mloc];
      }
    }
  }
}

// ---------------- combine: out = ys + yr[2t] + yr[2t+1] ----------------
__global__ __launch_bounds__(256) void moe_combine(
    float* __restrict__ out, const float* __restrict__ yr, int Ntok, int Dm) {
  int i = blockIdx.x * 256 + threadIdx.x;
  if (i >= Ntok * Dm) return;
  int t = i / Dm, c = i - t * Dm;
  out[i] += yr[(size_t)(2 * t) * Dm + c] + yr[(size_t)(2 * t + 1) * Dm + c];
}

// ---------------- host orchestration ----------------
extern "C" void kernel_launch(void* const* d_in, const int* in_sizes, int n_in,
                              void* d_out, int out_size, void* d_ws, size_t ws_size,
                              hipStream_t stream) {
  (void)n_in; (void)out_size; (void)ws_size;
  const float* x   = (const float*)d_in[0];
  const float* Ws1 = (const float*)d_in[1];
  const float* bs1 = (const float*)d_in[2];
  const float* Ws2 = (const float*)d_in[3];
  const float* bs2 = (const float*)d_in[4];
  const float* We1 = (const float*)d_in[5];
  const float* be1 = (const float*)d_in[6];
  const float* We2 = (const float*)d_in[7];
  const float* be2 = (const float*)d_in[8];
  const float* Wr  = (const float*)d_in[9];
  const float* br  = (const float*)d_in[10];
  float* out = (float*)d_out;
  const int Ntok = in_sizes[0] / D_MODEL;  // 2048

  // workspace layout (~294 MB)
  char* p = (char*)d_ws;
  int*    counts  = (int*)p;                 p += 256;
  int*    rowlist = (int*)p;                 p += (size_t)NEXP * Ntok * 4;
  float*  wlist   = (float*)p;               p += (size_t)NEXP * Ntok * 4;
  __bf16* xb      = (__bf16*)p;              p += (size_t)Ntok * D_MODEL * 2;
  __bf16* Ws1t    = (__bf16*)p;              p += (size_t)2 * HS_DIM * D_MODEL * 2;
  __bf16* Ws2t    = (__bf16*)p;              p += (size_t)D_MODEL * HS_DIM * 2;
  __bf16* We1t    = (__bf16*)p;              p += (size_t)NEXP * 2 * HR_DIM * D_MODEL * 2;
  __bf16* We2t    = (__bf16*)p;              p += (size_t)NEXP * D_MODEL * HR_DIM * 2;
  __bf16* actS    = (__bf16*)p;              p += (size_t)Ntok * HS_DIM * 2;
  __bf16* actR    = (__bf16*)p;              p += (size_t)2 * Ntok * HR_DIM * 2;
  float*  yrbuf   = (float*)p;

  hipMemsetAsync(d_ws, 0, 256, stream);  // zero expert counters

  // preprocessing: bf16 conversion + weight transposition
  cvt_bf16<<<(Ntok * D_MODEL / 4 + 255) / 256, 256, 0, stream>>>(
      x, xb, Ntok * D_MODEL);
  transpose_cvt<<<dim3(2 * HS_DIM / 32, D_MODEL / 32, 1), 256, 0, stream>>>(
      Ws1, Ws1t, D_MODEL, 2 * HS_DIM);
  transpose_cvt<<<dim3(D_MODEL / 32, HS_DIM / 32, 1), 256, 0, stream>>>(
      Ws2, Ws2t, HS_DIM, D_MODEL);
  transpose_cvt<<<dim3(2 * HR_DIM / 32, D_MODEL / 32, NEXP), 256, 0, stream>>>(
      We1, We1t, D_MODEL, 2 * HR_DIM);
  transpose_cvt<<<dim3(D_MODEL / 32, HR_DIM / 32, NEXP), 256, 0, stream>>>(
      We2, We2t, HR_DIM, D_MODEL);

  moe_router<<<(Ntok + 7) / 8, 256, 0, stream>>>(x, Wr, br, counts, rowlist,
                                                 wlist, Ntok);
  // shared expert
  moe_l1<<<dim3(HS_DIM / NT, (Ntok + MT - 1) / MT, 1), 256, 0, stream>>>(
      xb, Ws1t, bs1, actS, nullptr, nullptr, Ntok, D_MODEL, HS_DIM);
  moe_l2<<<dim3(D_MODEL / NT, (Ntok + MT - 1) / MT, 1), 256, 0, stream>>>(
      actS, Ws2t, bs2, out, nullptr, nullptr, nullptr, Ntok, HS_DIM, D_MODEL);
  // routed experts (sparse dispatch; tiles past count exit immediately)
  moe_l1<<<dim3(HR_DIM / NT, (Ntok + MT - 1) / MT, NEXP), 256, 0, stream>>>(
      xb, We1t, be1, actR, rowlist, counts, Ntok, D_MODEL, HR_DIM);
  moe_l2<<<dim3(D_MODEL / NT, (Ntok + MT - 1) / MT, NEXP), 256, 0, stream>>>(
      actR, We2t, be2, yrbuf, rowlist, counts, wlist, Ntok, HR_DIM, D_MODEL);
  moe_combine<<<(Ntok * D_MODEL + 255) / 256, 256, 0, stream>>>(out, yrbuf,
                                                               Ntok, D_MODEL);
}
